// GPT_1451698946272
// MI455X (gfx1250) — matrix operand order repaired and checked by reading, not compiled, not run on hardware
//
#include <hip/hip_runtime.h>
#include <stdint.h>

// ---------------- types / constants ----------------
typedef __attribute__((ext_vector_type(16))) __bf16 v16bf;
typedef __attribute__((ext_vector_type(8)))  float  v8f;

#define DD 768
#define FF 3072
#define VV 50257
#define TT 1024
#define BB 4
#define LL 12
#define NH 12
#define HDIM 64
#define MM (BB*TT)   // 4096 token rows

static __device__ __forceinline__ uint16_t f2bf(float f){
  uint32_t u = __float_as_uint(f);
  u += 0x7FFFu + ((u >> 16) & 1u);          // round-to-nearest-even
  return (uint16_t)(u >> 16);
}

union FragU { uint32_t u[8]; v16bf v; };

// A-matrix 16x32 bf16 fragment per ISA layout:
// lanes 0-15: M=lane, VGPR j<4 -> K=2j,2j+1 ; j>=4 -> K=16+2(j-4)
// lanes 16-31: same M, K ranges shifted by +8
static __device__ __forceinline__ v16bf load_frag_a(const uint16_t* rowbase, int k0, int lane){
  FragU f; int half = lane >> 4;
  #pragma unroll
  for (int j = 0; j < 8; ++j){
    int k = k0 + ((j < 4) ? (2*j) : (16 + 2*(j-4))) + 8*half;
    f.u[j] = *(const uint32_t*)(rowbase + k);
  }
  return f.v;
}

// B-matrix 32x16 bf16 fragment; source stored BT-layout [n][k] contiguous-k.
static __device__ __forceinline__ v16bf load_frag_b(const uint16_t* colbase, int k0, int lane){
  FragU f; int half = lane >> 4;
  #pragma unroll
  for (int j = 0; j < 8; ++j){
    int k = k0 + ((j < 4) ? (2*j) : (16 + 2*(j-4))) + 8*half;
    f.u[j] = *(const uint32_t*)(colbase + k);
  }
  return f.v;
}

static __device__ __forceinline__ v8f wmma_bf16(v16bf a, v16bf b, v8f c){
  return __builtin_amdgcn_wmma_f32_16x16x32_bf16(false, a, false, b, (short)0, c, false, false);
}

// async copy 16B/lane from global to LDS (ASYNCcnt-tracked)
static __device__ __forceinline__ void async_ld16(uint32_t lds_addr, const void* gptr){
  asm volatile("global_load_async_to_lds_b128 %0, %1, off"
               :: "v"(lds_addr), "v"((uint64_t)(size_t)gptr) : "memory");
}
static __device__ __forceinline__ void wait_async0(void){
  asm volatile("s_wait_asynccnt 0x0" ::: "memory");
}
static __device__ __forceinline__ void wait_async4(void){
  asm volatile("s_wait_asynccnt 0x4" ::: "memory");   // allow 4 in flight (next tile)
}
static __device__ __forceinline__ uint32_t lds_off(const void* p){
  return (uint32_t)(size_t)p;   // low 32 bits of generic ptr = LDS byte address
}

// ---------------- weight preprocessing ----------------
__global__ void cast_bf16_kernel(const float* __restrict__ in, uint16_t* __restrict__ out, size_t n){
  size_t i = (size_t)blockIdx.x*blockDim.x + threadIdx.x;
  size_t stride = (size_t)gridDim.x*blockDim.x;
  for (; i < n; i += stride) out[i] = f2bf(in[i]);
}

// in: [L][R][C] fp32  ->  out: [L][C][R] bf16   (BT layout per layer)
__global__ void transpose_cast_kernel(const float* __restrict__ in, uint16_t* __restrict__ out,
                                      int Lc, int R, int C){
  size_t n = (size_t)Lc*R*C;
  size_t i = (size_t)blockIdx.x*blockDim.x + threadIdx.x;
  size_t stride = (size_t)gridDim.x*blockDim.x;
  size_t rc = (size_t)R*C;
  for (; i < n; i += stride){
    size_t l = i / rc, rem = i - l*rc;
    size_t r = rem / C, c = rem - r*C;
    out[l*rc + c*(size_t)R + r] = f2bf(in[i]);
  }
}

// ---------------- embedding ----------------
__global__ __launch_bounds__(256) void embed_kernel(const int* __restrict__ tok,
                                                    const float* __restrict__ wte,
                                                    const float* __restrict__ wpe,
                                                    float* __restrict__ x){
  int row = blockIdx.x, tid = threadIdx.x;
  int t = row & (TT-1);
  int tk = tok[row];
  for (int c = tid; c < DD; c += 256)
    x[(size_t)row*DD + c] = wte[(size_t)tk*DD + c] + wpe[(size_t)t*DD + c];
}

// ---------------- layernorm -> bf16 ----------------
__global__ __launch_bounds__(256) void ln_bf16_kernel(const float* __restrict__ x,
                                                      const float* __restrict__ g,
                                                      const float* __restrict__ bta,
                                                      uint16_t* __restrict__ out){
  __shared__ float red[256];
  int row = blockIdx.x, tid = threadIdx.x;
  const float* xr = x + (size_t)row*DD;
  float s = 0.f;
  for (int c = tid; c < DD; c += 256) s += xr[c];
  red[tid] = s; __syncthreads();
  for (int o = 128; o > 0; o >>= 1){ if (tid < o) red[tid] += red[tid+o]; __syncthreads(); }
  float mean = red[0] * (1.0f/DD); __syncthreads();
  float v = 0.f;
  for (int c = tid; c < DD; c += 256){ float d = xr[c]-mean; v += d*d; }
  red[tid] = v; __syncthreads();
  for (int o = 128; o > 0; o >>= 1){ if (tid < o) red[tid] += red[tid+o]; __syncthreads(); }
  float rstd = rsqrtf(red[0]*(1.0f/DD) + 1e-5f);
  for (int c = tid; c < DD; c += 256)
    out[(size_t)row*DD + c] = f2bf((xr[c]-mean)*rstd*g[c] + bta[c]);
}

// ---------------- generic WMMA GEMM ----------------
// C[M,N] = act(A[M,K] * BT[N,K]^T + bias) (+ residual)  ; A,BT bf16 row-major
// 128x128 block tile, 8 wave32s, each wave 32x64 (2x4 WMMA tiles), K-step 32.
// Double-buffered async global->LDS pipeline: next tile's copies run under
// the barriers and WMMAs of the current tile.
__global__ __launch_bounds__(256) void gemm_bf16_kernel(
    const uint16_t* __restrict__ A, const uint16_t* __restrict__ BT,
    const float* __restrict__ bias, const float* __restrict__ residual,
    void* __restrict__ outp, int M, int N, int K, int gelu, int out_f32)
{
  __shared__ uint16_t sA[2][128*32];
  __shared__ uint16_t sB[2][128*32];
  const int tid = threadIdx.x, lane = tid & 31, wave = tid >> 5;
  const int lcol = lane & 15, group = lane >> 4;
  const int bm = blockIdx.y * 128, bn = blockIdx.x * 128;
  const int wm = (wave & 3) * 32, wn = (wave >> 2) * 64;

  // ---- per-thread staging coords: 2 chunks of 8 bf16 per matrix ----
  const int r0 = tid >> 2;            // rows 0..63 (chunk1 = +64)
  const int sg = (tid & 3) * 8;       // 8-element segment within 32-col tile
  const uint16_t* aG0 = A + (size_t)(bm + r0)*K + sg;
  const uint16_t* aG1 = aG0 + (size_t)64*K;
  int nb0 = bn + r0;      if (nb0 >= N) nb0 = N - 1;   // clamp (masked at store)
  int nb1 = bn + r0 + 64; if (nb1 >= N) nb1 = N - 1;
  const uint16_t* bG0 = BT + (size_t)nb0*K + sg;
  const uint16_t* bG1 = BT + (size_t)nb1*K + sg;
  const uint32_t aL0 = lds_off(&sA[0][r0*32 + sg]);   // buf1 = +8192 bytes
  const uint32_t bL0 = lds_off(&sB[0][r0*32 + sg]);

  auto issue_tile = [&](int bufi, int kk){
    const uint32_t o = bufi ? 8192u : 0u;
    async_ld16(aL0 + o,        aG0 + kk);
    async_ld16(aL0 + o + 4096, aG1 + kk);
    async_ld16(bL0 + o,        bG0 + kk);
    async_ld16(bL0 + o + 4096, bG1 + kk);
  };

  const v8f vzero = {0,0,0,0,0,0,0,0};
  v8f acc[2][4];
  #pragma unroll
  for (int i = 0; i < 2; ++i)
    #pragma unroll
    for (int j = 0; j < 4; ++j) acc[i][j] = vzero;

  issue_tile(0, 0);
  int cur = 0;
  for (int kk = 0; kk < K; kk += 32){
    const bool more = (kk + 32 < K);
    if (more){
      issue_tile(cur ^ 1, kk + 32);
      if (kk + 64 < K) __builtin_prefetch(aG0 + kk + 64, 0, 1);
      wait_async4();            // previous tile complete, next tile in flight
    } else {
      wait_async0();
    }
    __syncthreads();

    const uint16_t* cA = sA[cur];
    const uint16_t* cB = sB[cur];
    v16bf af0 = load_frag_a(cA + (wm +      lcol)*32, 0, lane);
    v16bf af1 = load_frag_a(cA + (wm + 16 + lcol)*32, 0, lane);
    v16bf bf0 = load_frag_b(cB + (wn +  0 + lcol)*32, 0, lane);
    v16bf bf1 = load_frag_b(cB + (wn + 16 + lcol)*32, 0, lane);
    v16bf bf2 = load_frag_b(cB + (wn + 32 + lcol)*32, 0, lane);
    v16bf bf3 = load_frag_b(cB + (wn + 48 + lcol)*32, 0, lane);

    acc[0][0] = wmma_bf16(af0, bf0, acc[0][0]);
    acc[0][1] = wmma_bf16(af0, bf1, acc[0][1]);
    acc[0][2] = wmma_bf16(af0, bf2, acc[0][2]);
    acc[0][3] = wmma_bf16(af0, bf3, acc[0][3]);
    acc[1][0] = wmma_bf16(af1, bf0, acc[1][0]);
    acc[1][1] = wmma_bf16(af1, bf1, acc[1][1]);
    acc[1][2] = wmma_bf16(af1, bf2, acc[1][2]);
    acc[1][3] = wmma_bf16(af1, bf3, acc[1][3]);
    __syncthreads();
    cur ^= 1;
  }

  // epilogue: C 16x16 layout -> row = r + 8*(lane>>4), col = lane&15
  #pragma unroll
  for (int mt = 0; mt < 2; ++mt)
    #pragma unroll
    for (int nt = 0; nt < 4; ++nt)
      #pragma unroll
      for (int r = 0; r < 8; ++r){
        int row = bm + wm + mt*16 + r + 8*group;
        int col = bn + wn + nt*16 + lcol;
        if (col < N){
          float v = acc[mt][nt][r];
          if (bias) v += bias[col];
          if (gelu) v = 0.5f*v*(1.0f + erff(v*0.70710678118f));
          size_t idx = (size_t)row*N + col;
          if (out_f32){
            float* o = (float*)outp;
            o[idx] = (residual ? residual[idx] : 0.0f) + v;
          } else {
            ((uint16_t*)outp)[idx] = f2bf(v);
          }
        }
      }
}

// ---------------- flash attention (causal, HD=64) ----------------
// grid: (T/64, B*NH), block 128 = 4 wave32s; wave w -> queries [qtile*64+16w, +16)
__global__ __launch_bounds__(128) void attn_kernel(const uint16_t* __restrict__ qkv,
                                                   uint16_t* __restrict__ out){
  __shared__ uint16_t sK[32*64];      // [key][hd]  == BT layout for Q*K^T
  __shared__ uint16_t sV[64*32];      // [hd][key]  == BT layout for P*V
  __shared__ uint16_t sP[4][16*32];   // per-wave P staging
  const int tid = threadIdx.x, lane = tid & 31, wave = tid >> 5;
  const int lcol = lane & 15, group = lane >> 4;
  const int qtile = blockIdx.x;
  const int b = blockIdx.y / NH, h = blockIdx.y % NH;
  const int q0 = qtile*64 + wave*16;

  const uint16_t* qbase = qkv + (size_t)(b*TT + q0 + lcol)*(3*DD) + h*HDIM;
  v16bf qa0 = load_frag_a(qbase, 0,  lane);
  v16bf qa1 = load_frag_a(qbase, 32, lane);

  // staging coords: 2 chunks of 8 bf16 per thread (keys 0..15 and 16..31)
  const int key0 = tid >> 3;          // 0..15
  const int sg   = (tid & 7) * 8;     // segment within 64-wide hd row
  const uint16_t* kG0 = qkv + (size_t)(b*TT + key0)*(3*DD) + DD + h*HDIM + sg;
  const uint32_t kL0 = lds_off(&sK[key0*64 + sg]);
  const uint32_t kL1 = kL0 + 16*64*2;

  const v8f vzero = {0,0,0,0,0,0,0,0};
  float rowmax[8], rowsum[8];
  v8f o[4];
  #pragma unroll
  for (int r = 0; r < 8; ++r){ rowmax[r] = -1e30f; rowsum[r] = 0.f; }
  #pragma unroll
  for (int nt = 0; nt < 4; ++nt) o[nt] = vzero;

  const int kend = (qtile+1)*64;
  for (int kb = 0; kb < kend; kb += 32){
    const uint16_t* kptr = kG0 + (size_t)kb*(3*DD);
    // K: async straight into LDS; V: load b128, scatter-transpose into sV
    async_ld16(kL0, kptr);
    async_ld16(kL1, kptr + (size_t)16*(3*DD));
    uint4 vv0 = *(const uint4*)(kptr + DD);
    uint4 vv1 = *(const uint4*)(kptr + (size_t)16*(3*DD) + DD);
    const uint16_t* pv0 = (const uint16_t*)&vv0;
    const uint16_t* pv1 = (const uint16_t*)&vv1;
    #pragma unroll
    for (int j = 0; j < 8; ++j){
      sV[(sg + j)*32 + key0]      = pv0[j];
      sV[(sg + j)*32 + key0 + 16] = pv1[j];
    }
    wait_async0();
    __syncthreads();

    if (kb <= q0 + 15){                      // wave-uniform skip of fully-masked blocks
      v8f s0 = vzero, s1 = vzero;
      const uint16_t* kb0 = sK + lcol*64;
      const uint16_t* kb1 = sK + (16 + lcol)*64;
      s0 = wmma_bf16(qa0, load_frag_b(kb0, 0,  lane), s0);
      s0 = wmma_bf16(qa1, load_frag_b(kb0, 32, lane), s0);
      s1 = wmma_bf16(qa0, load_frag_b(kb1, 0,  lane), s1);
      s1 = wmma_bf16(qa1, load_frag_b(kb1, 32, lane), s1);

      #pragma unroll
      for (int r = 0; r < 8; ++r){
        int qg = q0 + r + 8*group;
        float v0 = s0[r]*0.125f; if (kb + lcol      > qg) v0 = -1e30f;
        float v1 = s1[r]*0.125f; if (kb + 16 + lcol > qg) v1 = -1e30f;
        float mx = fmaxf(v0, v1);
        #pragma unroll
        for (int off = 1; off < 16; off <<= 1) mx = fmaxf(mx, __shfl_xor(mx, off, 16));
        float nm = fmaxf(rowmax[r], mx);
        float al = __expf(rowmax[r] - nm);
        rowmax[r] = nm;
        float p0 = __expf(v0 - nm), p1 = __expf(v1 - nm);
        float ps = p0 + p1;
        #pragma unroll
        for (int off = 1; off < 16; off <<= 1) ps += __shfl_xor(ps, off, 16);
        rowsum[r] = rowsum[r]*al + ps;
        o[0][r] *= al; o[1][r] *= al; o[2][r] *= al; o[3][r] *= al;
        int prow = r + 8*group;
        sP[wave][prow*32 + lcol]      = f2bf(p0);
        sP[wave][prow*32 + 16 + lcol] = f2bf(p1);
      }
      asm volatile("s_wait_dscnt 0x0" ::: "memory");   // wave-local LDS RAW
      v16bf pf = load_frag_a(&sP[wave][lcol*32], 0, lane);
      #pragma unroll
      for (int nt = 0; nt < 4; ++nt)
        o[nt] = wmma_bf16(pf, load_frag_b(sV + (nt*16 + lcol)*32, 0, lane), o[nt]);
    }
    __syncthreads();
  }

  #pragma unroll
  for (int r = 0; r < 8; ++r){
    float inv = 1.0f / rowsum[r];
    int row = q0 + r + 8*group;
    #pragma unroll
    for (int nt = 0; nt < 4; ++nt){
      int col = h*HDIM + nt*16 + lcol;
      out[(size_t)(b*TT + row)*DD + col] = f2bf(o[nt][r]*inv);
    }
  }
}

// ---------------- host orchestration ----------------
extern "C" void kernel_launch(void* const* d_in, const int* in_sizes, int n_in,
                              void* d_out, int out_size, void* d_ws, size_t ws_size,
                              hipStream_t stream)
{
  (void)in_sizes; (void)n_in; (void)out_size; (void)ws_size;
  const int*   tokens = (const int*)  d_in[0];
  const float* wte    = (const float*)d_in[1];
  const float* wpe    = (const float*)d_in[2];
  const float* ln1_g  = (const float*)d_in[3];
  const float* ln1_b  = (const float*)d_in[4];
  const float* attn_w = (const float*)d_in[5];
  const float* attn_b = (const float*)d_in[6];
  const float* apw    = (const float*)d_in[7];
  const float* apb    = (const float*)d_in[8];
  const float* ln2_g  = (const float*)d_in[9];
  const float* ln2_b  = (const float*)d_in[10];
  const float* fc_w   = (const float*)d_in[11];
  const float* fc_b   = (const float*)d_in[12];
  const float* mpw    = (const float*)d_in[13];
  const float* mpb    = (const float*)d_in[14];
  const float* lnf_g  = (const float*)d_in[15];
  const float* lnf_b  = (const float*)d_in[16];

  char* ws = (char*)d_ws;
  size_t off = 0;
  auto take = [&](size_t bytes)->char*{
    char* p = ws + off; off = (off + bytes + 255) & ~(size_t)255; return p;
  };
  uint16_t* wte_bf = (uint16_t*)take((size_t)VV*DD*2);          // [V][D]  (lm head BT)
  uint16_t* aw_t   = (uint16_t*)take((size_t)LL*DD*3*DD*2);     // [L][3D][D]
  uint16_t* ap_t   = (uint16_t*)take((size_t)LL*DD*DD*2);       // [L][D][D]
  uint16_t* fc_t   = (uint16_t*)take((size_t)LL*DD*FF*2);       // [L][F][D]
  uint16_t* mp_t   = (uint16_t*)take((size_t)LL*FF*DD*2);       // [L][D][F]
  float*    x      = (float*)   take((size_t)MM*DD*4);          // residual stream
  uint16_t* h_bf   = (uint16_t*)take((size_t)MM*DD*2);          // LN output
  uint16_t* qkv_bf = (uint16_t*)take((size_t)MM*3*DD*2);
  uint16_t* ao_bf  = (uint16_t*)take((size_t)MM*DD*2);
  uint16_t* mh_bf  = (uint16_t*)take((size_t)MM*FF*2);

  cast_bf16_kernel     <<<2048,256,0,stream>>>(wte, wte_bf, (size_t)VV*DD);
  transpose_cast_kernel<<<2048,256,0,stream>>>(attn_w, aw_t, LL, DD, 3*DD);
  transpose_cast_kernel<<<2048,256,0,stream>>>(apw,    ap_t, LL, DD, DD);
  transpose_cast_kernel<<<2048,256,0,stream>>>(fc_w,   fc_t, LL, DD, FF);
  transpose_cast_kernel<<<2048,256,0,stream>>>(mpw,    mp_t, LL, FF, DD);
  embed_kernel<<<MM,256,0,stream>>>(tokens, wte, wpe, x);

  for (int l = 0; l < LL; ++l){
    ln_bf16_kernel<<<MM,256,0,stream>>>(x, ln1_g + (size_t)l*DD, ln1_b + (size_t)l*DD, h_bf);
    gemm_bf16_kernel<<<dim3(3*DD/128, MM/128), 256, 0, stream>>>(
        h_bf, aw_t + (size_t)l*DD*3*DD, attn_b + (size_t)l*3*DD, nullptr,
        qkv_bf, MM, 3*DD, DD, 0, 0);
    attn_kernel<<<dim3(TT/64, BB*NH), 128, 0, stream>>>(qkv_bf, ao_bf);
    gemm_bf16_kernel<<<dim3(DD/128, MM/128), 256, 0, stream>>>(
        ao_bf, ap_t + (size_t)l*DD*DD, apb + (size_t)l*DD, x,
        x, MM, DD, DD, 0, 1);
    ln_bf16_kernel<<<MM,256,0,stream>>>(x, ln2_g + (size_t)l*DD, ln2_b + (size_t)l*DD, h_bf);
    gemm_bf16_kernel<<<dim3(FF/128, MM/128), 256, 0, stream>>>(
        h_bf, fc_t + (size_t)l*DD*FF, fc_b + (size_t)l*FF, nullptr,
        mh_bf, MM, FF, DD, 1, 0);
    gemm_bf16_kernel<<<dim3(DD/128, MM/128), 256, 0, stream>>>(
        mh_bf, mp_t + (size_t)l*FF*DD, mpb + (size_t)l*DD, x,
        x, MM, DD, FF, 0, 1);
  }
  ln_bf16_kernel<<<MM,256,0,stream>>>(x, lnf_g, lnf_b, h_bf);
  gemm_bf16_kernel<<<dim3((VV+127)/128, MM/128), 256, 0, stream>>>(
      h_bf, wte_bf, nullptr, nullptr, (float*)d_out, MM, VV, DD, 0, 1);
}
